// BAGNNConv_39367670235414
// MI455X (gfx1250) — matrix operand order repaired and checked by reading, not compiled, not run on hardware
//
#include <hip/hip_runtime.h>
#include <hip/hip_bf16.h>
#include <stdint.h>

// ---------------------------------------------------------------------------
// BAGNN heterogeneous GAT layer, restructured for MI455X (gfx1250, wave32):
//   * attention logits reduced to per-node scalar tables (thin bf16 WMMA GEMM)
//   * per-edge work reduced to scalar gathers + alpha*x scatter (f32 atomics)
//   * per-edge-type dense projection hoisted to ONE node-level bf16 WMMA GEMM
//   * all WMMA B-operands pre-swizzled into per-lane fragment order so each
//     lane fragment is a single contiguous 32B load; A loads are 128-bit.
// Compute: ~25 GFLOP bf16 WMMA (trivial). Traffic: ~2 GB => HBM-bound O(100us).
// ---------------------------------------------------------------------------

typedef __attribute__((ext_vector_type(16))) __bf16 v16bf;
typedef __attribute__((ext_vector_type(8)))  __bf16 v8bf;
typedef __attribute__((ext_vector_type(4)))  __bf16 v4bf;
typedef __attribute__((ext_vector_type(8)))  float  v8f;

// monotonic float->uint key so segment-max can use atomicMax on a zeroed buffer
__device__ __forceinline__ unsigned fkey(float f) {
  unsigned u = __float_as_uint(f);
  return (u & 0x80000000u) ? ~u : (u | 0x80000000u);
}
__device__ __forceinline__ float finv(unsigned k) {
  unsigned u = (k & 0x80000000u) ? (k & 0x7fffffffu) : ~k;
  return __uint_as_float(u);
}

// ---------------- Stage A1: tiny constants --------------------------------
__global__ void k_const(const float* __restrict__ Wb, const float* __restrict__ A,
                        const float* __restrict__ relW, const float* __restrict__ behW,
                        const float* __restrict__ a_att,
                        float* a0Wb, float* a1Wb, float* v0, float* v1,
                        float* cconst, float* bterm3) {
  int t = threadIdx.x; // 256 threads
  if (t < 128) {
    float s = 0.f;
    for (int i = 0; i < 128; ++i) s += a_att[i] * Wb[i * 128 + t];
    a0Wb[t] = s;
  } else {
    int j = t - 128;
    float s = 0.f;
    for (int i = 0; i < 128; ++i) s += a_att[128 + i] * Wb[i * 128 + j];
    a1Wb[j] = s;
  }
  if (t < 64) {
    int phi = t >> 4, r = t & 15;
    float s0 = 0.f, s1 = 0.f;
    for (int i = 0; i < 128; ++i) {
      float a = A[phi * 2048 + i * 16 + r];
      s0 += a_att[i] * a;
      s1 += a_att[128 + i] * a;
    }
    v0[t] = s0; v1[t] = s1;
  }
  if (t < 10) {
    float s = 0.f;
    for (int i = 0; i < 128; ++i) s += relW[t * 128 + i] * a_att[256 + i];
    if (t < 6) {
      int b = t % 3;
      for (int i = 0; i < 128; ++i) s += behW[b * 128 + i] * a_att[384 + i];
    }
    cconst[t] = s;
  }
  if (t < 3) {
    float s = 0.f;
    for (int i = 0; i < 128; ++i) s += behW[t * 128 + i] * a_att[384 + i];
    bterm3[t] = s;
  }
}

// ---------------- Stage A2: bf16 operand matrices, fragment-swizzled ------
// Swizzled layout for a KxNcols B operand: element e of lane l for tile
// (colTile ct, kStep kb) lives at ((ct*(K/32)+kb)*32 + l)*16 + e and holds
// Bm[k][col] with k = kb*32 + (l>>4)*16 + e, col = ct*16 + (l&15).
// Logical values:
//   U[t] (128x16): per-node-type scalar-projection vectors (see round 0)
//   Wt[et] (128x128) = W_et^T, Wtb = W_base^T, At[p] (32x128) = A[p+1]^T pad.
__device__ __forceinline__ float u_val(const float* B, const float* a0Wb,
                                       const float* a1Wb, const float* v0,
                                       const float* v1, int t, int col, int j) {
  int phi0 = -1, phi1 = -1, b = -1; bool base1 = false;
  if (t == 0)      { if (col < 3) { phi0 = 0; b = col; } else if (col < 6) { phi1 = 1; b = col - 3; } }
  else if (t == 1) { if (col < 3) { phi0 = 1; b = col; } else if (col < 6) { phi1 = 0; b = col - 3; } else if (col == 6) base1 = true; }
  else             { if (col < 3) { phi0 = (t == 2) ? 2 : 3; b = col; } else if (col == 3) base1 = true; }
  float val = 0.f;
  if (phi0 >= 0) {
    val = a0Wb[j];
    for (int r = 0; r < 16; ++r) val += B[b * 2048 + j * 16 + r] * v0[phi0 * 16 + r];
  } else if (phi1 >= 0) {
    val = a1Wb[j];
    for (int r = 0; r < 16; ++r) val += B[b * 2048 + j * 16 + r] * v1[phi1 * 16 + r];
  } else if (base1) val = a1Wb[j];
  return val;
}

__global__ void k_prep(const float* __restrict__ Wb, const float* __restrict__ A,
                       const float* __restrict__ B,
                       const float* __restrict__ a0Wb, const float* __restrict__ a1Wb,
                       const float* __restrict__ v0, const float* __restrict__ v1,
                       __bf16* __restrict__ U, __bf16* __restrict__ Wt,
                       __bf16* __restrict__ Wtb, __bf16* __restrict__ At) {
  int idx = blockIdx.x * blockDim.x + threadIdx.x;
  if (idx < 8192) {                       // U tables: 4 types, 1 colTile, K=128
    int t = idx >> 11, rem = idx & 2047;
    int e = rem & 15, lane = (rem >> 4) & 31, kb = rem >> 9;
    int k = kb * 32 + (lane >> 4) * 16 + e;
    int col = lane & 15;
    U[idx] = (__bf16)u_val(B, a0Wb, a1Wb, v0, v1, t, col, k);
    return;
  }
  idx -= 8192;
  if (idx < 98304) {                      // Wt[et]: 8 colTiles, K=128
    int et = idx / 16384, rem = idx & 16383;
    int e = rem & 15, lane = (rem >> 4) & 31, kb = (rem >> 9) & 3, ct = rem >> 11;
    int k = kb * 32 + (lane >> 4) * 16 + e;
    int col = ct * 16 + (lane & 15);
    int phi = (et < 3) ? 0 : 1, b = et % 3;
    float val = Wb[col * 128 + k];        // W_et[col][k]
    for (int r = 0; r < 16; ++r) val += A[phi * 2048 + col * 16 + r] * B[b * 2048 + k * 16 + r];
    Wt[et * 16384 + (rem ^ 0)] = (__bf16)val;
    return;
  }
  idx -= 98304;
  if (idx < 16384) {                      // Wtb
    int rem = idx;
    int e = rem & 15, lane = (rem >> 4) & 31, kb = (rem >> 9) & 3, ct = rem >> 11;
    int k = kb * 32 + (lane >> 4) * 16 + e;
    int col = ct * 16 + (lane & 15);
    Wtb[rem] = (__bf16)Wb[col * 128 + k];
    return;
  }
  idx -= 16384;
  if (idx < 12288) {                      // At (phi = p+1): 8 colTiles, K=32
    int p = idx / 4096, rem = idx & 4095;
    int e = rem & 15, lane = (rem >> 4) & 31, ct = rem >> 9;
    int k = (lane >> 4) * 16 + e;
    int col = ct * 16 + (lane & 15);
    float val = (k < 16) ? A[(p + 1) * 2048 + col * 16 + k] : 0.f;
    At[p * 4096 + rem] = (__bf16)val;
  }
}

// pack two contiguous float4 (8 floats) into the low/high 8-halfs of a frag
__device__ __forceinline__ void cvt8(v16bf& a, int base, float4 f0, float4 f1) {
  a[base + 0] = (__bf16)f0.x; a[base + 1] = (__bf16)f0.y;
  a[base + 2] = (__bf16)f0.z; a[base + 3] = (__bf16)f0.w;
  a[base + 4] = (__bf16)f1.x; a[base + 5] = (__bf16)f1.y;
  a[base + 6] = (__bf16)f1.z; a[base + 7] = (__bf16)f1.w;
}

// ---------------- thin GEMM: S = X(Nx128) @ U(128x16), bf16 WMMA ----------
__global__ void k_gemm_s(const float* __restrict__ X, int Nrows,
                         const __bf16* __restrict__ Us, float* __restrict__ S) {
  int wave = threadIdx.x >> 5, lane = threadIdx.x & 31;
  int tile = blockIdx.x * 8 + wave;
  int row0 = tile * 16;
  if (row0 >= Nrows) return;              // uniform per wave
  int r = lane & 15, hi = lane >> 4;
  int row = row0 + r;
  row = row < Nrows ? row : (Nrows - 1);  // clamp: tail rows never stored
  int koffA = hi * 16, col = r;
  const float4* Xr = reinterpret_cast<const float4*>(X + (size_t)row * 128);
  v8f acc = {};
#pragma unroll
  for (int kb = 0; kb < 4; ++kb) {
    v16bf a;
    int q = (kb * 32 + koffA) >> 2;       // float4 index of low segment
    cvt8(a, 0, Xr[q], Xr[q + 1]);
    cvt8(a, 8, Xr[q + 2], Xr[q + 3]);     // +8 floats = +2 float4
    v16bf bm = *reinterpret_cast<const v16bf*>(Us + ((size_t)kb * 32 + lane) * 16);
    acc = __builtin_amdgcn_wmma_f32_16x16x32_bf16(false, a, false, bm, (short)0, acc, false, false);
  }
#pragma unroll
  for (int g = 0; g < 8; ++g) {
    int orow = row0 + hi * 8 + g;
    if (orow < Nrows) S[(size_t)orow * 16 + col] = acc[g];
  }
}

// ---------------- wide GEMM: Out(Nx128) += Asrc(NxK) @ Bs(swizzled Kx128) --
template <int K>
__global__ void k_gemm_w(const float* __restrict__ Asrc, int Nrows,
                         const __bf16* __restrict__ Bs, float* __restrict__ Out) {
  __shared__ alignas(16) __bf16 sA[16 * K];
  int row0 = blockIdx.x * 16;
  // stage 16 x K A-tile into LDS as bf16 (vector loads/stores, clamped rows)
  for (int idx4 = threadIdx.x; idx4 < 16 * K / 4; idx4 += 256) {
    int r = idx4 / (K / 4), k4 = idx4 - r * (K / 4);
    int row = row0 + r;
    row = row < Nrows ? row : (Nrows - 1);
    float4 f = reinterpret_cast<const float4*>(Asrc + (size_t)row * K)[k4];
    v4bf h; h[0] = (__bf16)f.x; h[1] = (__bf16)f.y; h[2] = (__bf16)f.z; h[3] = (__bf16)f.w;
    *reinterpret_cast<v4bf*>(&sA[r * K + k4 * 4]) = h;
  }
  __syncthreads();
  int wave = threadIdx.x >> 5, lane = threadIdx.x & 31;
  int r = lane & 15, hi = lane >> 4;
  int koffA = hi * 16;
  int col = wave * 16 + r;
  v8f acc = {};
#pragma unroll
  for (int kb = 0; kb < K / 32; ++kb) {
    v8bf lo = *reinterpret_cast<const v8bf*>(&sA[r * K + kb * 32 + koffA]);
    v8bf hi8 = *reinterpret_cast<const v8bf*>(&sA[r * K + kb * 32 + 8 + koffA]);
    v16bf a = __builtin_shufflevector(lo, hi8, 0, 1, 2, 3, 4, 5, 6, 7,
                                      8, 9, 10, 11, 12, 13, 14, 15);
    v16bf bm = *reinterpret_cast<const v16bf*>(
        Bs + (((size_t)wave * (K / 32) + kb) * 32 + lane) * 16);
    acc = __builtin_amdgcn_wmma_f32_16x16x32_bf16(false, a, false, bm, (short)0, acc, false, false);
  }
#pragma unroll
  for (int g = 0; g < 8; ++g) {
    int orow = row0 + hi * 8 + g;
    if (orow < Nrows) Out[(size_t)orow * 128 + col] += acc[g];
  }
}

// ---------------- per-edge pass 1: logits + segment max -------------------
__global__ void k_edge_score(const int* __restrict__ ei, int E,
                             const float* __restrict__ Ss, const float* __restrict__ Sd,
                             int cs, int cd, const float* __restrict__ cconst, int et,
                             const int* __restrict__ attr, const float* __restrict__ bterm3,
                             float* __restrict__ ebuf, unsigned* __restrict__ mkey) {
  int i = blockIdx.x * blockDim.x + threadIdx.x;
  if (i >= E) return;
  int s = ei[i], d = ei[E + i];
  float cc = cconst[et];
  int c = cs;
  if (attr) {
    int o = attr[i]; o = o < 0 ? 0 : (o > 2 ? 2 : o);
    c = o; cc += bterm3[o];
  }
  float e = Ss[(size_t)s * 16 + c] + Sd[(size_t)d * 16 + cd] + cc;
  ebuf[i] = e;
  atomicMax(&mkey[d], fkey(e));
}

// ---------------- per-edge pass 2: exp + segment sum ----------------------
__global__ void k_edge_exp(const int* __restrict__ ei, int E,
                           float* __restrict__ ebuf,
                           const unsigned* __restrict__ mkey, float* __restrict__ ssum) {
  int i = blockIdx.x * blockDim.x + threadIdx.x;
  if (i >= E) return;
  int d = ei[E + i];
  float ex = __expf(ebuf[i] - finv(mkey[d]));
  ebuf[i] = ex;
  atomicAdd(&ssum[d], ex);
}

// ---------------- per-edge pass 3: scatter alpha*x (wave per edge) --------
__global__ void k_edge_scatter(const int* __restrict__ ei, int E,
                               const float* __restrict__ X, const float* __restrict__ ebuf,
                               const float* __restrict__ ssum,
                               float* __restrict__ T, float* __restrict__ Tmid,
                               const float* __restrict__ Bmat, const int* __restrict__ attr) {
  int wave = threadIdx.x >> 5, lane = threadIdx.x & 31;
  int i = blockIdx.x * 8 + wave;
  if (i >= E) return;                     // uniform per wave
  int s = ei[i], d = ei[E + i];
  float alpha = ebuf[i] / (ssum[d] + 1e-16f);
  const float4 xv = reinterpret_cast<const float4*>(X + (size_t)s * 128)[lane];
  float* td = T + (size_t)d * 128 + lane * 4;
  atomicAdd(td + 0, alpha * xv.x);
  atomicAdd(td + 1, alpha * xv.y);
  atomicAdd(td + 2, alpha * xv.z);
  atomicAdd(td + 3, alpha * xv.w);
  if (Tmid) {                              // structural: mid = x @ B[origin]
    int o = attr[i]; o = o < 0 ? 0 : (o > 2 ? 2 : o);
    const float* Bo = Bmat + o * 2048 + (lane * 4) * 16;
    for (int r = 0; r < 16; ++r) {
      float p = xv.x * Bo[r] + xv.y * Bo[16 + r] + xv.z * Bo[32 + r] + xv.w * Bo[48 + r];
      for (int off = 16; off > 0; off >>= 1) p += __shfl_down(p, off, 32);
      if (lane == 0) atomicAdd(&Tmid[(size_t)d * 32 + r], alpha * p);
    }
  }
}

// ---------------- finalize: LayerNorm + residual + ELU (wave per row) -----
__global__ void k_final(float* __restrict__ Out,
                        const float* __restrict__ xu, const float* __restrict__ xp,
                        const float* __restrict__ xc, const float* __restrict__ xb,
                        const float* __restrict__ gamma, const float* __restrict__ beta,
                        int totalRows) {
  int wave = threadIdx.x >> 5, lane = threadIdx.x & 31;
  int g = blockIdx.x * 8 + wave;
  if (g >= totalRows) return;
  const float* xrow;
  if (g < 100000)      xrow = xu + (size_t)g * 128;
  else if (g < 200000) xrow = xp + (size_t)(g - 100000) * 128;
  else if (g < 201000) xrow = xc + (size_t)(g - 200000) * 128;
  else                 xrow = xb + (size_t)(g - 201000) * 128;
  float* orow = Out + (size_t)g * 128;
  float4 h = reinterpret_cast<float4*>(orow)[lane];
  float s = h.x + h.y + h.z + h.w;
  for (int off = 16; off > 0; off >>= 1) s += __shfl_xor(s, off, 32);
  float mu = s * (1.f / 128.f);
  float v = (h.x - mu) * (h.x - mu) + (h.y - mu) * (h.y - mu) +
            (h.z - mu) * (h.z - mu) + (h.w - mu) * (h.w - mu);
  for (int off = 16; off > 0; off >>= 1) v += __shfl_xor(v, off, 32);
  float rstd = rsqrtf(v * (1.f / 128.f) + 1e-5f);
  float4 gm = reinterpret_cast<const float4*>(gamma)[lane];
  float4 bt = reinterpret_cast<const float4*>(beta)[lane];
  float4 xr = reinterpret_cast<const float4*>(xrow)[lane];
  float4 y;
  y.x = (h.x - mu) * rstd * gm.x + bt.x + xr.x;
  y.y = (h.y - mu) * rstd * gm.y + bt.y + xr.y;
  y.z = (h.z - mu) * rstd * gm.z + bt.z + xr.z;
  y.w = (h.w - mu) * rstd * gm.w + bt.w + xr.w;
  y.x = y.x > 0.f ? y.x : (__expf(y.x) - 1.f);
  y.y = y.y > 0.f ? y.y : (__expf(y.y) - 1.f);
  y.z = y.z > 0.f ? y.z : (__expf(y.z) - 1.f);
  y.w = y.w > 0.f ? y.w : (__expf(y.w) - 1.f);
  reinterpret_cast<float4*>(orow)[lane] = y;
}

// ---------------------------------------------------------------------------
extern "C" void kernel_launch(void* const* d_in, const int* in_sizes, int n_in,
                              void* d_out, int out_size, void* d_ws, size_t ws_size,
                              hipStream_t stream) {
  const float* x[4] = {(const float*)d_in[0], (const float*)d_in[1],
                       (const float*)d_in[2], (const float*)d_in[3]};
  const int    N[4]      = {100000, 100000, 1000, 2000};
  const size_t rowOff[4] = {0, 100000, 200000, 201000};
  const float* Wb    = (const float*)d_in[18];
  const float* A     = (const float*)d_in[19];
  const float* B     = (const float*)d_in[20];
  const float* relW  = (const float*)d_in[21];
  const float* behW  = (const float*)d_in[22];
  const float* a_att = (const float*)d_in[23];
  const float* gam   = (const float*)d_in[24];
  const float* bet   = (const float*)d_in[25];
  (void)n_in; (void)ws_size;

  // bump allocator over d_ws (~80 MB total)
  uintptr_t p = (uintptr_t)d_ws;
  auto alloc = [&](size_t bytes) -> void* {
    uintptr_t q = (p + 255) & ~(uintptr_t)255; p = q + bytes; return (void*)q;
  };
  float*    a0Wb   = (float*)alloc(128 * 4);
  float*    a1Wb   = (float*)alloc(128 * 4);
  float*    v0     = (float*)alloc(64 * 4);
  float*    v1     = (float*)alloc(64 * 4);
  float*    cconst = (float*)alloc(16 * 4);
  float*    bterm3 = (float*)alloc(4 * 4);
  __bf16*   U      = (__bf16*)alloc(8192 * 2);
  __bf16*   Wt     = (__bf16*)alloc((size_t)6 * 16384 * 2);
  __bf16*   Wtb    = (__bf16*)alloc(16384 * 2);
  __bf16*   At     = (__bf16*)alloc((size_t)3 * 4096 * 2);
  float*    S[4];
  for (int t = 0; t < 4; ++t) S[t] = (float*)alloc((size_t)N[t] * 16 * 4);
  float*    ebuf = (float*)alloc(100000 * 4);
  unsigned* mkey = (unsigned*)alloc(100000 * 4);
  float*    ssum = (float*)alloc(100000 * 4);
  float*    T    = (float*)alloc((size_t)100000 * 128 * 4);
  float*    Tmid = (float*)alloc((size_t)100000 * 32 * 4);

  float* agg = (float*)d_out;
  (void)hipMemsetAsync(d_out, 0, (size_t)out_size * sizeof(float), stream);

  k_const<<<1, 256, 0, stream>>>(Wb, A, relW, behW, a_att, a0Wb, a1Wb, v0, v1, cconst, bterm3);
  k_prep<<<(135168 + 255) / 256, 256, 0, stream>>>(Wb, A, B, a0Wb, a1Wb, v0, v1, U, Wt, Wtb, At);
  for (int t = 0; t < 4; ++t) {
    int tiles = (N[t] + 15) / 16;
    k_gemm_s<<<(tiles + 7) / 8, 256, 0, stream>>>(x[t], N[t], U + t * 2048, S[t]);
  }

  // edge-type table: {ei input idx, attr idx(-1=behavioral), src_t, dst_t, cs, cd, phi}
  struct ET { int ei, attr, st, dt, cs, cd, phi; };
  const ET ets[10] = {
    { 4, -1, 0, 1, 0, 3, -1}, { 5, -1, 0, 1, 1, 4, -1}, { 6, -1, 0, 1, 2, 5, -1},
    { 7, -1, 1, 0, 0, 3, -1}, { 8, -1, 1, 0, 1, 4, -1}, { 9, -1, 1, 0, 2, 5, -1},
    {10, 11, 1, 2, -1, 3, 1}, {12, 13, 2, 1, -1, 6, 2},
    {14, 15, 1, 3, -1, 3, 1}, {16, 17, 3, 1, -1, 6, 3}
  };
  for (int e = 0; e < 10; ++e) {
    const ET& m = ets[e];
    int E = in_sizes[m.ei] / 2;
    const int* ei   = (const int*)d_in[m.ei];
    const int* attr = (m.attr >= 0) ? (const int*)d_in[m.attr] : nullptr;
    int nd = N[m.dt];
    (void)hipMemsetAsync(mkey, 0, (size_t)nd * 4, stream);
    (void)hipMemsetAsync(ssum, 0, (size_t)nd * 4, stream);
    (void)hipMemsetAsync(T,    0, (size_t)nd * 128 * 4, stream);
    if (attr) (void)hipMemsetAsync(Tmid, 0, (size_t)nd * 32 * 4, stream);
    int nb = (E + 255) / 256;
    k_edge_score<<<nb, 256, 0, stream>>>(ei, E, S[m.st], S[m.dt], m.cs, m.cd,
                                         cconst, e, attr, bterm3, ebuf, mkey);
    k_edge_exp<<<nb, 256, 0, stream>>>(ei, E, ebuf, mkey, ssum);
    k_edge_scatter<<<(E + 7) / 8, 256, 0, stream>>>(ei, E, x[m.st], ebuf, ssum,
                                                    T, attr ? Tmid : nullptr, B, attr);
    float* outT = agg + rowOff[m.dt] * 128;
    if (attr) {
      k_gemm_w<128><<<(nd + 15) / 16, 256, 0, stream>>>(T, nd, Wtb, outT);
      k_gemm_w<32><<<(nd + 15) / 16, 256, 0, stream>>>(Tmid, nd, At + (size_t)(m.phi - 1) * 4096, outT);
    } else {
      k_gemm_w<128><<<(nd + 15) / 16, 256, 0, stream>>>(T, nd, Wt + (size_t)e * 16384, outT);
    }
  }

  const int totalRows = 203000;
  k_final<<<(totalRows + 7) / 8, 256, 0, stream>>>(agg, x[0], x[1], x[2], x[3], gam, bet, totalRows);
}
